// Generator_54692113547279
// MI455X (gfx1250) — matrix operand order repaired and checked by reading, not compiled, not run on hardware
//
#include <hip/hip_runtime.h>

// ---------------------------------------------------------------------------
// StyleGAN2-style generator forward for gfx1250 (MI455X), WMMA implicit-GEMM
// ---------------------------------------------------------------------------

typedef __attribute__((ext_vector_type(16))) _Float16 v16h;
typedef __attribute__((ext_vector_type(8)))  float    v8f;
typedef int v4i_t __attribute__((vector_size(16)));    // int4 for async-LDS builtin

#define SQRT2F 1.41421356237309515f
#define INV_SQRT512 0.044194173824159216f

#if defined(__has_builtin)
#if __has_builtin(__builtin_amdgcn_global_load_async_to_lds_b128)
#define HAVE_ASYNC_B128 1
#endif
#endif

#define AS_GLOBAL __attribute__((address_space(1)))
#define AS_LDS    __attribute__((address_space(3)))

__device__ __forceinline__ float lrelu_f(float v) { return v > 0.f ? v : 0.2f * v; }
__device__ __forceinline__ int clampi(int v, int lo, int hi) {
  return v < lo ? lo : (v > hi ? hi : v);
}

#ifdef HAVE_ASYNC_B128
__device__ __forceinline__ void wait_async0() {
#if __has_builtin(__builtin_amdgcn_s_wait_asynccnt)
  __builtin_amdgcn_s_wait_asynccnt(0);
#else
  asm volatile("s_wait_asynccnt 0" ::: "memory");
#endif
}
#endif

// 4x4 blur kernel = sum of four shifted [1 2 1; 2 4 2; 1 2 1]/16 kernels
__device__ __constant__ float c_blur[16] = {
  0.0625f, 0.1875f, 0.1875f, 0.0625f,
  0.1875f, 0.5625f, 0.5625f, 0.1875f,
  0.1875f, 0.5625f, 0.5625f, 0.1875f,
  0.0625f, 0.1875f, 0.1875f, 0.0625f
};

// ---------------------------------------------------------------------------
// Mapping network
// ---------------------------------------------------------------------------
__global__ void k_pixelnorm(const float* __restrict__ z, float* __restrict__ out) {
  __shared__ float red[256];
  const int n = blockIdx.x, t = threadIdx.x;
  float s = 0.f;
  for (int i = t; i < 512; i += 256) { float v = z[n * 512 + i]; s += v * v; }
  red[t] = s; __syncthreads();
  for (int k = 128; k > 0; k >>= 1) { if (t < k) red[t] += red[t + k]; __syncthreads(); }
  const float inv = rsqrtf(red[0] / 512.f + 1e-8f);
  for (int i = t; i < 512; i += 256) out[n * 512 + i] = z[n * 512 + i] * inv;
}

__global__ void k_fc_lrelu(const float* __restrict__ in, const float* __restrict__ w,
                           const float* __restrict__ b, float* __restrict__ out) {
  const int o = blockIdx.x * blockDim.x + threadIdx.x;
  const int n = blockIdx.y;
  if (o >= 512) return;
  float acc = 0.f;
  for (int i = 0; i < 512; ++i) acc += in[n * 512 + i] * w[o * 512 + i];
  const float v = (acc * (0.01f * INV_SQRT512) + b[o] * 0.01f) * SQRT2F;
  out[n * 512 + o] = lrelu_f(v);
}

// mod[n,i] = eq_fc(trunc_style, sw, sb, lr=1) + 1
__global__ void k_mod(const float* __restrict__ wlat, const float* __restrict__ avg,
                      const float* __restrict__ sw, const float* __restrict__ sb,
                      float* __restrict__ mod, int iC, int styleIdx) {
  const int i = blockIdx.x * blockDim.x + threadIdx.x;
  const int n = blockIdx.y;
  if (i >= iC) return;
  const float rate = (styleIdx < 10) ? 0.7f : 1.0f;
  float acc = 0.f;
  for (int k = 0; k < 512; ++k) {
    const float s = avg[k] + (wlat[n * 512 + k] - avg[k]) * rate;
    acc += s * sw[i * 512 + k];
  }
  mod[n * iC + i] = acc * INV_SQRT512 + sb[i] + 1.0f;
}

// ssum[o,i] = (1/(iC*9)) * sum_kl w^2
__global__ void k_ssum(const float* __restrict__ w, float* __restrict__ ssum,
                       int iC, int oC, int isU) {
  const int idx = blockIdx.x * blockDim.x + threadIdx.x;
  if (idx >= oC * iC) return;
  const int o = idx / iC, i = idx % iC;
  const float sc2 = 1.0f / (float)(iC * 9);
  float s = 0.f;
  for (int t = 0; t < 9; ++t) {
    const float r = isU ? w[(i * oC + o) * 9 + t] : w[(o * iC + i) * 9 + t];
    s += r * r;
  }
  ssum[idx] = s * sc2;
}

__global__ void k_demod(const float* __restrict__ ssum, const float* __restrict__ mod,
                        float* __restrict__ d, int iC, int oC) {
  const int idx = blockIdx.x * blockDim.x + threadIdx.x;
  if (idx >= 2 * oC) return;
  const int n = idx / oC, o = idx % oC;
  float acc = 0.f;
  for (int i = 0; i < iC; ++i) {
    const float m = mod[n * iC + i];
    acc += ssum[o * iC + i] * m * m;
  }
  d[idx] = rsqrtf(acc + 1e-8f);
}

// Pack weights K-major f16: wpk[k*oC+o], k=(i*9+kh*3+kw). Flip+transpose for convT.
__global__ void k_pack(const float* __restrict__ w, _Float16* __restrict__ wpk,
                       int iC, int oC, int isU) {
  const int idx = blockIdx.x * blockDim.x + threadIdx.x;
  const int K = iC * 9;
  if (idx >= K * oC) return;
  const int k = idx / oC, o = idx % oC;
  const int i = k / 9, rr = k % 9, kh = rr / 3, kw = rr % 3;
  const float sc = rsqrtf((float)(iC * 9));
  const float v = isU ? w[(i * oC + o) * 9 + (2 - kh) * 3 + (2 - kw)]
                      : w[(o * iC + i) * 9 + rr];
  wpk[idx] = (_Float16)(v * sc);
}

__global__ void k_broadcast(const float* __restrict__ src, float* __restrict__ dst, int per) {
  const int idx = blockIdx.x * blockDim.x + threadIdx.x;
  if (idx >= 2 * per) return;
  dst[idx] = src[idx % per];
}

// ---------------------------------------------------------------------------
// WMMA implicit-GEMM modulated conv.
// mode 0: 3x3 pad-1 conv.  mode 1: conv_transpose stride-2 (lhs-dilated gather).
// fullEpi 1: *demod + noise + bias + lrelu*sqrt2.  fullEpi 0: *demod only.
// Block = 128 threads (4 waves). Block M-tile = 32 positions; wave w owns
// columns [blockIdx.y*64 + 16w, +16) and issues 2 WMMAs per K=32 slab
// (two M sub-tiles sharing one B fragment).
// K = iC*9 is a multiple of 32 for all layers; oC, nbase multiples of 16.
// ---------------------------------------------------------------------------
__global__ __launch_bounds__(128)
void k_conv_wmma(const float* __restrict__ x, const _Float16* __restrict__ wpk,
                 const float* __restrict__ mod, const float* __restrict__ demod,
                 const float* __restrict__ noise, const float* __restrict__ nscale,
                 const float* __restrict__ bias, float* __restrict__ out,
                 int iC, int oC, int H, int W, int OH, int OW, int mode, int fullEpi) {
  __shared__ __align__(16) _Float16 As[32][34];       // [k][m], 32 K x 32 M (+pad)
  __shared__ __align__(16) _Float16 Bs[4][32][16];    // per-wave [k][ncol], 32B rows

  const int tid  = threadIdx.x;
  const int wave = tid >> 5;
  const int lane = tid & 31;
  const int n    = blockIdx.z;
  const int M    = OH * OW;
  const int K    = iC * 9;
  const int mbase = blockIdx.x * 32;
  const int nbase = blockIdx.y * 64 + wave * 16;
  const bool waveActive = (nbase < oC);

  v8f acc0 = {};
  v8f acc1 = {};

  for (int k0 = 0; k0 < K; k0 += 32) {
    // ---- stage B: per-wave 32(K) x 16(N) weight tile -----------------------
    if (waveActive) {
#ifdef HAVE_ASYNC_B128
      // lane -> (row, 8-half column group); two issues cover rows 0..31.
      {
        const int row0 = lane >> 1;
        const int col8 = (lane & 1) * 8;
        const _Float16* g0 = wpk + (size_t)(k0 + row0) * oC + nbase + col8;
        const _Float16* g1 = wpk + (size_t)(k0 + row0 + 16) * oC + nbase + col8;
        __builtin_amdgcn_global_load_async_to_lds_b128(
            (AS_GLOBAL v4i_t*)(void*)g0,
            (AS_LDS v4i_t*)(void*)&Bs[wave][row0][col8], 0, 0);
        __builtin_amdgcn_global_load_async_to_lds_b128(
            (AS_GLOBAL v4i_t*)(void*)g1,
            (AS_LDS v4i_t*)(void*)&Bs[wave][row0 + 16][col8], 0, 0);
      }
#else
      {
        const int kk = k0 + lane;
        for (int c = 0; c < 16; ++c)
          Bs[wave][lane][c] = wpk[(size_t)kk * oC + nbase + c];
      }
#endif
    }

    // prefetch next weight slab (global_prefetch_b8)
    if (k0 + 32 < K)
      __builtin_prefetch((const void*)(wpk + (size_t)(k0 + 32) * oC + (waveActive ? nbase : 0)), 0, 1);

    // ---- stage A: 32(K) x 32(M) modulated/gathered input, f32 -> f16 -------
    for (int s = 0; s < 8; ++s) {
      const int q = tid + 128 * s;            // 1024 entries / 128 threads
      const int k = q >> 5, m = q & 31;
      float v = 0.f;
      const int kk = k0 + k;                  // always < K (K % 32 == 0)
      const int pos = mbase + m;
      if (pos < M) {
        const int i = kk / 9, rr = kk % 9, kh = rr / 3, kw = rr % 3;
        const int oh = pos / OW, ow = pos % OW;
        if (mode == 0) {
          const int ih = oh + kh - 1, iw = ow + kw - 1;
          if (ih >= 0 && ih < H && iw >= 0 && iw < W)
            v = x[((size_t)(n * iC + i) * H + ih) * W + iw] * mod[n * iC + i];
        } else {
          const int a = oh + kh - 2, b = ow + kw - 2;   // dilated-input coords
          if (a >= 0 && b >= 0 && !(a & 1) && !(b & 1)) {
            const int ih = a >> 1, iw = b >> 1;
            if (ih < H && iw < W)
              v = x[((size_t)(n * iC + i) * H + ih) * W + iw] * mod[n * iC + i];
          }
        }
      }
      As[k][m] = (_Float16)v;
    }

#ifdef HAVE_ASYNC_B128
    wait_async0();                            // B tile landed in LDS
#endif
    __syncthreads();

    // ---- build fragments per ISA wave32 16-bit layouts; 2 WMMAs ------------
    if (waveActive) {
      v16h af0, af1, bf;
      const int hi  = (lane >= 16) ? 1 : 0;
      const int row = lane & 15;              // A row (M) / B col (N)
#pragma unroll
      for (int j = 0; j < 16; ++j) {
        const int ka = (j < 8 ? j : j + 8) + hi * 8;   // A: K {0..7,16..23}/{8..15,24..31}
        af0[j] = As[ka][row];
        af1[j] = As[ka][row + 16];
        const int kb = (j < 8 ? j : j + 8) + hi * 8;   // B: K {0..7,16..23}/{8..15,24..31}
        bf[j] = Bs[wave][kb][row];
      }
      acc0 = __builtin_amdgcn_wmma_f32_16x16x32_f16(false, af0, false, bf,
                                                    (short)0, acc0, false, false);
      acc1 = __builtin_amdgcn_wmma_f32_16x16x32_f16(false, af1, false, bf,
                                                    (short)0, acc1, false, false);
    }
    __syncthreads();
  }

  // ---- epilogue: C layout — lane<16: N=lane, M=r; lane>=16: N=lane-16, M=r+8
  if (waveActive) {
    const int hi   = (lane >= 16) ? 8 : 0;
    const int ncol = lane & 15;
    const int o    = nbase + ncol;
    if (o < oC) {
      const float dmul = demod ? demod[n * oC + o] : 1.0f;
      const float ns   = (fullEpi && nscale) ? nscale[0] : 0.0f;
#pragma unroll
      for (int t = 0; t < 2; ++t) {
        const v8f& a = t ? acc1 : acc0;
#pragma unroll
        for (int r = 0; r < 8; ++r) {
          const int pos = mbase + t * 16 + r + hi;
          if (pos < M) {
            const int oh = pos / OW, ow = pos % OW;
            float v = a[r] * dmul;
            if (fullEpi) {
              v += ns * noise[pos];
              v = lrelu_f((v + bias[o]) * SQRT2F);
            }
            out[((size_t)(n * oC + o) * OH + oh) * OW + ow] = v;
          }
        }
      }
    }
  }
}

// ---------------------------------------------------------------------------
// Depthwise 4x4 blur (pad 1), fused noise+bias+lrelu epilogue. S=2H+1 -> O=2H
// ---------------------------------------------------------------------------
__global__ void k_blur(const float* __restrict__ t, const float* __restrict__ noise,
                       const float* __restrict__ nsc, const float* __restrict__ bias,
                       float* __restrict__ out, int C, int S, int O) {
  const long long idx = (long long)blockIdx.x * blockDim.x + threadIdx.x;
  const long long total = 2LL * C * O * O;
  if (idx >= total) return;
  const int ow = (int)(idx % O);
  const int oh = (int)((idx / O) % O);
  const int c  = (int)((idx / ((long long)O * O)) % C);
  const int n  = (int)(idx / ((long long)O * O * C));
  const float* tp = t + ((size_t)(n * C + c)) * S * S;
  float acc = 0.f;
#pragma unroll
  for (int a = 0; a < 4; ++a) {
    const int ih = oh + a - 1;
    if (ih < 0 || ih >= S) continue;
#pragma unroll
    for (int b = 0; b < 4; ++b) {
      const int iw = ow + b - 1;
      if (iw < 0 || iw >= S) continue;
      acc += c_blur[a * 4 + b] * tp[ih * S + iw];
    }
  }
  acc += nsc[0] * noise[oh * O + ow];
  out[idx] = lrelu_f((acc + bias[c]) * SQRT2F);
}

// ---------------------------------------------------------------------------
// toRGB: 1x1 modulated conv (no demod) + bias
// ---------------------------------------------------------------------------
__global__ void k_torgb(const float* __restrict__ x, const float* __restrict__ w,
                        const float* __restrict__ mod, const float* __restrict__ bias,
                        float* __restrict__ out, int iC, int HW) {
  const long long idx = (long long)blockIdx.x * blockDim.x + threadIdx.x;
  if (idx >= 2LL * HW) return;
  const int n = (int)(idx / HW), pix = (int)(idx % HW);
  const float sc = rsqrtf((float)iC);
  float a0 = 0.f, a1 = 0.f, a2 = 0.f;
  for (int i = 0; i < iC; ++i) {
    const float xv = x[((size_t)(n * iC + i)) * HW + pix] * mod[n * iC + i];
    a0 += w[i] * xv;
    a1 += w[iC + i] * xv;
    a2 += w[2 * iC + i] * xv;
  }
  out[((size_t)(n * 3 + 0)) * HW + pix] = a0 * sc + bias[0];
  out[((size_t)(n * 3 + 1)) * HW + pix] = a1 * sc + bias[1];
  out[((size_t)(n * 3 + 2)) * HW + pix] = a2 * sc + bias[2];
}

// ---------------------------------------------------------------------------
// Bilinear 2x upsample (half-pixel centers, edge clamp) accumulated into dst
// ---------------------------------------------------------------------------
__global__ void k_upadd(const float* __restrict__ src, float* __restrict__ dst,
                        int S, int O) {
  const long long idx = (long long)blockIdx.x * blockDim.x + threadIdx.x;
  const long long total = 2LL * 3 * O * O;
  if (idx >= total) return;
  const int ow = (int)(idx % O);
  const int oh = (int)((idx / O) % O);
  const int c  = (int)((idx / ((long long)O * O)) % 3);
  const int n  = (int)(idx / ((long long)O * O * 3));
  const float fy = (oh + 0.5f) * 0.5f - 0.5f;
  const float fx = (ow + 0.5f) * 0.5f - 0.5f;
  const int y0 = (int)floorf(fy), x0 = (int)floorf(fx);
  const float ty = fy - (float)y0, tx = fx - (float)x0;
  const int y0c = clampi(y0, 0, S - 1), y1c = clampi(y0 + 1, 0, S - 1);
  const int x0c = clampi(x0, 0, S - 1), x1c = clampi(x0 + 1, 0, S - 1);
  const float* sp = src + ((size_t)(n * 3 + c)) * S * S;
  const float v = (1.f - ty) * ((1.f - tx) * sp[y0c * S + x0c] + tx * sp[y0c * S + x1c])
                + ty * ((1.f - tx) * sp[y1c * S + x0c] + tx * sp[y1c * S + x1c]);
  dst[idx] += v;
}

// ---------------------------------------------------------------------------
// Host orchestration
// ---------------------------------------------------------------------------
extern "C" void kernel_launch(void* const* d_in, const int* in_sizes, int n_in,
                              void* d_out, int out_size, void* d_ws, size_t ws_size,
                              hipStream_t stream) {
  (void)in_sizes; (void)n_in; (void)out_size; (void)ws_size;
  const int N = 2;

  const float* z = (const float*)d_in[0];
  const float* map_w[8]; const float* map_b[8];
  for (int i = 0; i < 8; ++i) { map_w[i] = (const float*)d_in[1 + i]; map_b[i] = (const float*)d_in[9 + i]; }
  const float* avg = (const float*)d_in[17];
  const float* cin = (const float*)d_in[18];
  const float* conv_w[17]; const float* st_w[17]; const float* st_b[17];
  const float* nsc[17];    const float* noi[17];  const float* cb[17];
  for (int i = 0; i < 17; ++i) {
    conv_w[i] = (const float*)d_in[19 + i];
    st_w[i]   = (const float*)d_in[36 + i];
    st_b[i]   = (const float*)d_in[53 + i];
    nsc[i]    = (const float*)d_in[70 + i];
    noi[i]    = (const float*)d_in[87 + i];
    cb[i]     = (const float*)d_in[104 + i];
  }
  const float* rgb_w[9]; const float* rgb_sw[9]; const float* rgb_sb[9]; const float* rgb_b[9];
  for (int j = 0; j < 9; ++j) {
    rgb_w[j]  = (const float*)d_in[121 + j];
    rgb_sw[j] = (const float*)d_in[130 + j];
    rgb_sb[j] = (const float*)d_in[139 + j];
    rgb_b[j]  = (const float*)d_in[148 + j];
  }

  // workspace carve (bump allocator)
  char* p = (char*)d_ws;
  auto alloc = [&](size_t bytes) -> void* {
    void* r = (void*)p;
    p += (bytes + 255) & ~(size_t)255;
    return r;
  };
  float*    w0   = (float*)alloc((size_t)N * 512 * sizeof(float));
  float*    w1   = (float*)alloc((size_t)N * 512 * sizeof(float));
  float*    modb = (float*)alloc((size_t)N * 512 * sizeof(float));
  float*    ssum = (float*)alloc((size_t)512 * 512 * sizeof(float));
  float*    dem  = (float*)alloc((size_t)N * 512 * sizeof(float));
  _Float16* wpk  = (_Float16*)alloc((size_t)4608 * 512 * sizeof(_Float16));
  const size_t actmax = (size_t)N * 32 * 1024 * 1024;          // 64M floats
  float* xA   = (float*)alloc(actmax * sizeof(float));
  float* xB   = (float*)alloc(actmax * sizeof(float));
  float* tmpT = (float*)alloc((size_t)N * 32 * 1025 * 1025 * sizeof(float));
  float* sA   = (float*)alloc((size_t)N * 3 * 512 * 512 * sizeof(float));
  float* sB   = (float*)alloc((size_t)N * 3 * 512 * 512 * sizeof(float));

  // ---- mapping network ----
  k_pixelnorm<<<dim3(N), dim3(256), 0, stream>>>(z, w0);
  float* cw = w0; float* nw = w1;
  for (int l = 0; l < 8; ++l) {
    k_fc_lrelu<<<dim3(2, N), dim3(256), 0, stream>>>(cw, map_w[l], map_b[l], nw);
    float* t = cw; cw = nw; nw = t;
  }
  const float* wlat = cw;

  // ---- broadcast const input ----
  {
    const int per = 512 * 16;
    k_broadcast<<<dim3((2 * per + 255) / 256), dim3(256), 0, stream>>>(cin, xA, per);
  }
  float* cur = xA; float* nxt = xB;

  static const int CHh[9]  = {512, 512, 512, 512, 512, 256, 128, 64, 32};
  static const int RESh[9] = {4, 8, 16, 32, 64, 128, 256, 512, 1024};
  int liC[17], loC[17], lU[17], lires[17], lores[17];
  liC[0] = 512; loC[0] = 512; lU[0] = 0; lires[0] = 4; lores[0] = 4;
  for (int r = 1; r <= 8; ++r) {
    liC[2 * r - 1] = CHh[r - 1]; loC[2 * r - 1] = CHh[r]; lU[2 * r - 1] = 1;
    lires[2 * r - 1] = RESh[r - 1]; lores[2 * r - 1] = RESh[r];
    liC[2 * r] = CHh[r]; loC[2 * r] = CHh[r]; lU[2 * r] = 0;
    lires[2 * r] = RESh[r]; lores[2 * r] = RESh[r];
  }

  auto run_conv = [&](int li) {
    const int iC = liC[li], oC = loC[li], isU = lU[li];
    const int Hin = lires[li], Rout = lores[li];
    k_mod<<<dim3((iC + 255) / 256, N), dim3(256), 0, stream>>>(
        wlat, avg, st_w[li], st_b[li], modb, iC, li);
    k_ssum<<<dim3((oC * iC + 255) / 256), dim3(256), 0, stream>>>(conv_w[li], ssum, iC, oC, isU);
    k_demod<<<dim3((N * oC + 255) / 256), dim3(256), 0, stream>>>(ssum, modb, dem, iC, oC);
    const int K = iC * 9;
    k_pack<<<dim3((K * oC + 255) / 256), dim3(256), 0, stream>>>(conv_w[li], wpk, iC, oC, isU);
    if (!isU) {
      const int OH = Rout, M = OH * OH;
      dim3 g((M + 31) / 32, (oC + 63) / 64, N);
      k_conv_wmma<<<g, dim3(128), 0, stream>>>(cur, wpk, modb, dem, noi[li], nsc[li], cb[li],
                                               nxt, iC, oC, Hin, Hin, OH, OH, 0, 1);
    } else {
      const int S = 2 * Hin + 1, M = S * S;
      dim3 g((M + 31) / 32, (oC + 63) / 64, N);
      k_conv_wmma<<<g, dim3(128), 0, stream>>>(cur, wpk, modb, dem, nullptr, nullptr, nullptr,
                                               tmpT, iC, oC, Hin, Hin, S, S, 1, 0);
      const int O = Rout;
      const long long tot = 2LL * oC * O * O;
      k_blur<<<dim3((unsigned)((tot + 255) / 256)), dim3(256), 0, stream>>>(
          tmpT, noi[li], nsc[li], cb[li], nxt, oC, S, O);
    }
    float* t = cur; cur = nxt; nxt = t;
  };

  auto run_rgb = [&](int j, float* target) {
    const int iC = CHh[j], R = RESh[j], HW = R * R;
    const int sidx = (j == 0) ? 1 : (2 * j + 1);
    k_mod<<<dim3((iC + 255) / 256, N), dim3(256), 0, stream>>>(
        wlat, avg, rgb_sw[j], rgb_sb[j], modb, iC, sidx);
    const long long tot = 2LL * HW;
    k_torgb<<<dim3((unsigned)((tot + 255) / 256)), dim3(256), 0, stream>>>(
        cur, rgb_w[j], modb, rgb_b[j], target, iC, HW);
  };

  run_conv(0);
  float* prevSkip = sA;
  run_rgb(0, sA);
  for (int r = 0; r < 8; ++r) {
    run_conv(2 * r + 1);
    run_conv(2 * r + 2);
    float* target = (r == 7) ? (float*)d_out : ((prevSkip == sA) ? sB : sA);
    run_rgb(r + 1, target);
    const int O = RESh[r + 1], S = RESh[r];
    const long long tot = 2LL * 3 * O * O;
    k_upadd<<<dim3((unsigned)((tot + 255) / 256)), dim3(256), 0, stream>>>(prevSkip, target, S, O);
    prevSkip = target;
  }
}
